// MultiHeadAttention_38087769981307
// MI455X (gfx1250) — compile-verified, matrix operand order checked by reading, not run on hardware
//
#include <hip/hip_runtime.h>

// Problem constants (match reference setup_inputs)
#define B_   4
#define C_   768
#define T_   2048
#define H_   12
#define D_   64

typedef __attribute__((ext_vector_type(16))) __bf16 v16bf;
typedef __attribute__((ext_vector_type(8)))  __bf16 v8bf;
typedef __attribute__((ext_vector_type(8)))  float  v8f;

// fp32 -> bf16; plain cast lets the backend pick the native cvt op.
static __device__ __forceinline__ __bf16 f2bf(float f) { return (__bf16)f; }

// Load a 16x32 A-fragment (or [N][K]-major B-fragment) row for this lane.
// rowp points at the lane's row (K-contiguous); kb = (lane>=16 ? 8 : 0).
// Lane holds K = {kb..kb+7, kb+16..kb+23} as two 16B vector loads.
static __device__ __forceinline__ v16bf frag_bf16(const __bf16* rowp, int kb) {
  v8bf lo = *reinterpret_cast<const v8bf*>(rowp + kb);
  v8bf hi = *reinterpret_cast<const v8bf*>(rowp + kb + 16);
  v16bf r;
#pragma unroll
  for (int i = 0; i < 8; ++i) { r[i] = lo[i]; r[i + 8] = hi[i]; }
  return r;
}

static __device__ __forceinline__ v8f wmma_bf16(v16bf a, v16bf b, v8f c) {
  // (neg_a, A, neg_b, B, c_mod, C, reuse_a, reuse_b)
  return __builtin_amdgcn_wmma_f32_16x16x32_bf16(false, a, false, b, (short)0, c,
                                                 false, false);
}

// Fragment set for a 4M x 2N wave tile, and its load / MACC steps.
struct Frags { v16bf a[4]; v16bf b[2]; };

static __device__ __forceinline__ void load_frags(
    Frags& f, const __bf16* const (&arow)[4], const __bf16* const (&brow)[2],
    int k0, int kb) {
#pragma unroll
  for (int ms = 0; ms < 4; ++ms) f.a[ms] = frag_bf16(arow[ms] + k0, kb);
#pragma unroll
  for (int ns = 0; ns < 2; ++ns) f.b[ns] = frag_bf16(brow[ns] + k0, kb);
}

static __device__ __forceinline__ void do_wmma(v8f (&acc)[4][2], const Frags& f) {
#pragma unroll
  for (int ms = 0; ms < 4; ++ms)
#pragma unroll
    for (int ns = 0; ns < 2; ++ns)
      acc[ms][ns] = wmma_bf16(f.a[ms], f.b[ns], acc[ms][ns]);
}

// Ping-pong k-loop over K=C_ in steps of 32; no buffer-rotation moves.
static __device__ __forceinline__ void gemm_k_loop(
    v8f (&acc)[4][2], const __bf16* const (&arow)[4],
    const __bf16* const (&brow)[2], int kb) {
  Frags f0, f1;
  load_frags(f0, arow, brow, 0, kb);
#pragma unroll 1
  for (int k0 = 0; k0 < C_ - 64; k0 += 64) {
    load_frags(f1, arow, brow, k0 + 32, kb);
    do_wmma(acc, f0);
    load_frags(f0, arow, brow, k0 + 64, kb);
    do_wmma(acc, f1);
  }
  load_frags(f1, arow, brow, C_ - 32, kb);
  do_wmma(acc, f0);
  do_wmma(acc, f1);
}

// ---------------------------------------------------------------------------
// Prep A: flat fp32 -> bf16 convert (for Wqkv, Wout). n8 = elements/8.
// ---------------------------------------------------------------------------
__global__ __launch_bounds__(256) void cvt_f32_bf16_kernel(
    const float* __restrict__ src, __bf16* __restrict__ dst, int n8) {
  int i = blockIdx.x * blockDim.x + threadIdx.x;
  if (i >= n8) return;
  const float4 a = reinterpret_cast<const float4*>(src)[2 * i];
  const float4 b = reinterpret_cast<const float4*>(src)[2 * i + 1];
  v8bf o;
  o[0] = f2bf(a.x); o[1] = f2bf(a.y); o[2] = f2bf(a.z); o[3] = f2bf(a.w);
  o[4] = f2bf(b.x); o[5] = f2bf(b.y); o[6] = f2bf(b.z); o[7] = f2bf(b.w);
  reinterpret_cast<v8bf*>(dst)[i] = o;
}

// ---------------------------------------------------------------------------
// Prep B: x [B,C,T] fp32 -> xT [B,T,C] bf16 (64x64 tile transpose via LDS).
// ---------------------------------------------------------------------------
__global__ __launch_bounds__(256) void xpose_kernel(
    const float* __restrict__ x, __bf16* __restrict__ xT) {
  __shared__ __bf16 lds[64][72];
  const int tid = threadIdx.x;
  const int b  = blockIdx.z;
  const int t0 = blockIdx.x * 64;
  const int c0 = blockIdx.y * 64;
#pragma unroll
  for (int i = 0; i < 4; ++i) {
    int idx = tid + 256 * i;
    int row = idx >> 4;   // c_local
    int q   = idx & 15;   // t quad
    const float4 f = *reinterpret_cast<const float4*>(
        x + ((size_t)b * C_ + (c0 + row)) * T_ + t0 + q * 4);
    lds[q * 4 + 0][row] = f2bf(f.x);
    lds[q * 4 + 1][row] = f2bf(f.y);
    lds[q * 4 + 2][row] = f2bf(f.z);
    lds[q * 4 + 3][row] = f2bf(f.w);
  }
  __syncthreads();
#pragma unroll
  for (int i = 0; i < 2; ++i) {
    int chunk = tid + 256 * i;       // 512 chunks = 64 rows x 8
    int row = chunk >> 3;            // t_local
    int c8  = chunk & 7;
    v8bf v = *reinterpret_cast<const v8bf*>(&lds[row][c8 * 8]);
    *reinterpret_cast<v8bf*>(
        xT + ((size_t)b * T_ + t0 + row) * C_ + c0 + c8 * 8) = v;
  }
}

// ---------------------------------------------------------------------------
// Kernel 1: QKV projection, all-bf16 operands, no LDS.
// Block: 256 thr (8 waves) computes a 128(M=o) x 128(N=t) tile;
// each wave: 4 M-subtiles x 2 N-subtiles, ping-pong k-loop.
// Writes Q,K as bf16 [B,H,T,d] and V as bf16 [B,H,d,T].
// ---------------------------------------------------------------------------
__global__ __launch_bounds__(256) void qkv_proj_kernel(
    const __bf16* __restrict__ xT, const __bf16* __restrict__ Wb,
    const float* __restrict__ bqkv, __bf16* __restrict__ qT,
    __bf16* __restrict__ kT, __bf16* __restrict__ vB) {
  const int tid  = threadIdx.x;
  const int wave = tid >> 5, lane = tid & 31, half = lane >> 4, n = lane & 15;
  const int b     = blockIdx.z;
  const int Nbase = blockIdx.x * 128;   // t
  const int Mbase = blockIdx.y * 128;   // o feature (0..2303)
  const int n0    = (wave & 3) * 2;     // two N sub-tiles per wave
  const int m0    = (wave >> 2) * 4;    // four M sub-tiles per wave
  const int kb    = half * 8;

  const __bf16* arow[4];
  const __bf16* brow[2];
#pragma unroll
  for (int ms = 0; ms < 4; ++ms)
    arow[ms] = Wb + (size_t)(Mbase + (m0 + ms) * 16 + n) * C_;
#pragma unroll
  for (int ns = 0; ns < 2; ++ns)
    brow[ns] = xT + ((size_t)b * T_ + Nbase + (n0 + ns) * 16 + n) * C_;

  v8f acc[4][2];
#pragma unroll
  for (int i = 0; i < 4; ++i)
#pragma unroll
    for (int j = 0; j < 2; ++j)
#pragma unroll
      for (int e = 0; e < 8; ++e) acc[i][j][e] = 0.0f;

  gemm_k_loop(acc, arow, brow, kb);

  // Epilogue. 128-aligned M-tiles never cross the q/k/v split (768 % 128 == 0),
  // and 16-row subtiles never cross a head (64) boundary -> hoist routing.
  const int which  = Mbase / C_;        // 0=q, 1=k, 2=v (scalar)
  const int wibase = Mbase - which * C_;
#pragma unroll
  for (int ms = 0; ms < 4; ++ms) {
    const int sub = wibase + (m0 + ms) * 16;  // 0..767, 16-aligned
    const int h   = sub >> 6;
    const int dd0 = (sub & 63) + half * 8;    // multiple of 8
    const int fb  = Mbase + (m0 + ms) * 16 + half * 8;  // bias base (uniform)
#pragma unroll
    for (int ns = 0; ns < 2; ++ns) {
      const int t = Nbase + (n0 + ns) * 16 + n;
      if (which < 2) {
        // rows of this lane are consecutive dd -> one 16B store
        __bf16* dst = (which == 0 ? qT : kT) +
                      (((size_t)b * H_ + h) * T_ + t) * D_ + dd0;
        v8bf o;
#pragma unroll
        for (int r = 0; r < 8; ++r) o[r] = f2bf(acc[ms][ns][r] + bqkv[fb + r]);
        *reinterpret_cast<v8bf*>(dst) = o;
      } else {
#pragma unroll
        for (int r = 0; r < 8; ++r)
          vB[(((size_t)b * H_ + h) * D_ + dd0 + r) * T_ + t] =
              f2bf(acc[ms][ns][r] + bqkv[fb + r]);
      }
    }
  }
}

// ---------------------------------------------------------------------------
// Kernel 2: flash attention. Block: 128 thr (4 waves), each wave owns 16 rows
// of a 64-row q-tile for one (b,h); s-loop in 64-column tiles with online
// softmax. K frags batch-loaded before S WMMAs; V frags issued before the
// softmax VALU work so HBM latency hides behind exp/shfl.
// ---------------------------------------------------------------------------
__global__ __launch_bounds__(128) void attn_kernel(
    const __bf16* __restrict__ qT, const __bf16* __restrict__ kT,
    const __bf16* __restrict__ vB, const int* __restrict__ mask,
    __bf16* __restrict__ obuf) {
  __shared__ __bf16 ldsP[4][16][72];  // per-wave P tile (C-layout -> A-layout)

  const int tid  = threadIdx.x;
  const int wave = tid >> 5, lane = tid & 31, half = lane >> 4, n = lane & 15;
  const int b  = blockIdx.z, h = blockIdx.y;
  const int t0 = blockIdx.x * 64;
  const float scale = 0.125f;  // 1/sqrt(64)

  const __bf16* qrow = qT + (((size_t)b * H_ + h) * T_ + (t0 + wave * 16 + n)) * D_;
  v16bf qa[2];
#pragma unroll
  for (int ks = 0; ks < 2; ++ks) qa[ks] = frag_bf16(qrow, ks * 32 + half * 8);

  const __bf16* kbase = kT + ((size_t)b * H_ + h) * T_ * D_;
  const __bf16* vrow[4];
#pragma unroll
  for (int dt = 0; dt < 4; ++dt)
    vrow[dt] = vB + (((size_t)b * H_ + h) * D_ + dt * 16 + n) * T_;

  v8f oacc[4];
  float mst[8], lst[8];
#pragma unroll
  for (int i = 0; i < 4; ++i)
#pragma unroll
    for (int j = 0; j < 8; ++j) oacc[i][j] = 0.0f;
#pragma unroll
  for (int r = 0; r < 8; ++r) { mst[r] = -3.0e38f; lst[r] = 0.0f; }

  for (int s0 = 0; s0 < T_; s0 += 64) {
    // Batch-load all K fragments for this s-tile (one clause).
    v16bf kf[4][2];
#pragma unroll
    for (int nt = 0; nt < 4; ++nt) {
      const __bf16* krow = kbase + (size_t)(s0 + nt * 16 + n) * D_;
#pragma unroll
      for (int ks = 0; ks < 2; ++ks)
        kf[nt][ks] = frag_bf16(krow, ks * 32 + half * 8);
    }

    // S = scale * Q K^T
    v8f sfrag[4];
#pragma unroll
    for (int nt = 0; nt < 4; ++nt) {
      v8f s;
#pragma unroll
      for (int j = 0; j < 8; ++j) s[j] = 0.0f;
      s = wmma_bf16(qa[0], kf[nt][0], s);
      s = wmma_bf16(qa[1], kf[nt][1], s);
      int mv = mask[b * T_ + s0 + nt * 16 + n];  // column-wise mask
#pragma unroll
      for (int j = 0; j < 8; ++j)
        s[j] = (mv == 0) ? -10000.0f : (s[j] * scale);
      sfrag[nt] = s;
    }

    // Issue V fragment loads now; they are independent of the softmax.
    v16bf vf[4][2];
#pragma unroll
    for (int dt = 0; dt < 4; ++dt)
#pragma unroll
      for (int ks = 0; ks < 2; ++ks)
        vf[dt][ks] = frag_bf16(vrow[dt] + s0, ks * 32 + half * 8);

    // Row max: rows live across the 16-lane half -> shfl_xor within half.
    float nm[8], rsum[8], alpha[8];
#pragma unroll
    for (int r = 0; r < 8; ++r) {
      float mx = sfrag[0][r];
#pragma unroll
      for (int nt = 1; nt < 4; ++nt) mx = fmaxf(mx, sfrag[nt][r]);
#pragma unroll
      for (int off = 8; off >= 1; off >>= 1)
        mx = fmaxf(mx, __shfl_xor(mx, off, 32));
      nm[r] = fmaxf(mst[r], mx);
      rsum[r] = 0.0f;
    }

    // P = exp(S - m_new): accumulate row sums, stage P in LDS as bf16.
#pragma unroll
    for (int nt = 0; nt < 4; ++nt) {
#pragma unroll
      for (int r = 0; r < 8; ++r) {
        float p = __expf(sfrag[nt][r] - nm[r]);
        rsum[r] += p;
        ldsP[wave][r + half * 8][nt * 16 + n] = f2bf(p);
      }
    }
#pragma unroll
    for (int r = 0; r < 8; ++r) {
#pragma unroll
      for (int off = 8; off >= 1; off >>= 1)
        rsum[r] += __shfl_xor(rsum[r], off, 32);
      alpha[r] = __expf(mst[r] - nm[r]);
      lst[r]   = lst[r] * alpha[r] + rsum[r];
      mst[r]   = nm[r];
    }
#pragma unroll
    for (int dt = 0; dt < 4; ++dt)
#pragma unroll
      for (int r = 0; r < 8; ++r) oacc[dt][r] *= alpha[r];

    __syncthreads();  // P stores -> P loads
    // O += P * V   (A = P from LDS, B = V [d][s] K-major)
#pragma unroll
    for (int ks = 0; ks < 2; ++ks) {
      v16bf pa = frag_bf16(&ldsP[wave][n][0], ks * 32 + half * 8);
#pragma unroll
      for (int dt = 0; dt < 4; ++dt)
        oacc[dt] = wmma_bf16(pa, vf[dt][ks], oacc[dt]);
    }
    __syncthreads();  // protect LDS reuse next iteration
  }

  // Normalize and store o as bf16 [B,T,C] (K-major operand for out-proj).
  const int trow0 = t0 + wave * 16;
#pragma unroll
  for (int r = 0; r < 8; ++r) {
    float inv = 1.0f / lst[r];
    int t = trow0 + r + half * 8;
#pragma unroll
    for (int dt = 0; dt < 4; ++dt) {
      int c = h * D_ + dt * 16 + n;
      obuf[((size_t)b * T_ + t) * C_ + c] = f2bf(oacc[dt][r] * inv);
    }
  }
}

// ---------------------------------------------------------------------------
// Kernel 3: output projection, all-bf16 operands, no LDS.
// Block: 256 thr computes 128(M) x 128(N); 4 M x 2 N subtiles per wave,
// ping-pong k-loop. fp32 coalesced stores + bias.
// ---------------------------------------------------------------------------
__global__ __launch_bounds__(256) void out_proj_kernel(
    const __bf16* __restrict__ obuf, const __bf16* __restrict__ Wb,
    const float* __restrict__ bout, float* __restrict__ out) {
  const int tid  = threadIdx.x;
  const int wave = tid >> 5, lane = tid & 31, half = lane >> 4, n = lane & 15;
  const int b     = blockIdx.z;
  const int Nbase = blockIdx.x * 128;  // t
  const int Mbase = blockIdx.y * 128;  // out feature
  const int n0    = (wave & 3) * 2;
  const int m0    = (wave >> 2) * 4;
  const int kb    = half * 8;

  const __bf16* arow[4];
  const __bf16* brow[2];
#pragma unroll
  for (int ms = 0; ms < 4; ++ms)
    arow[ms] = Wb + (size_t)(Mbase + (m0 + ms) * 16 + n) * C_;
#pragma unroll
  for (int ns = 0; ns < 2; ++ns)
    brow[ns] = obuf + ((size_t)b * T_ + Nbase + (n0 + ns) * 16 + n) * C_;

  v8f acc[4][2];
#pragma unroll
  for (int i = 0; i < 4; ++i)
#pragma unroll
    for (int j = 0; j < 2; ++j)
#pragma unroll
      for (int e = 0; e < 8; ++e) acc[i][j][e] = 0.0f;

  gemm_k_loop(acc, arow, brow, kb);

#pragma unroll
  for (int ms = 0; ms < 4; ++ms) {
#pragma unroll
    for (int ns = 0; ns < 2; ++ns) {
      const int t = Nbase + (n0 + ns) * 16 + n;
#pragma unroll
      for (int r = 0; r < 8; ++r) {
        int m = Mbase + (m0 + ms) * 16 + r + half * 8;
        out[((size_t)b * C_ + m) * T_ + t] = acc[ms][ns][r] + bout[m];
      }
    }
  }
}

// ---------------------------------------------------------------------------
extern "C" void kernel_launch(void* const* d_in, const int* in_sizes, int n_in,
                              void* d_out, int out_size, void* d_ws, size_t ws_size,
                              hipStream_t stream) {
  (void)in_sizes; (void)n_in; (void)out_size; (void)ws_size;
  const float* x    = (const float*)d_in[0];
  const int*   mask = (const int*)d_in[1];
  const float* Wqkv = (const float*)d_in[2];
  const float* bqkv = (const float*)d_in[3];
  const float* Wout = (const float*)d_in[4];
  const float* bout = (const float*)d_in[5];
  float* out = (float*)d_out;

  // Workspace layout (bf16 elements). obuf aliases xT (disjoint lifetimes).
  const size_t NW_QKV = (size_t)3 * C_ * C_;
  const size_t NW_OUT = (size_t)C_ * C_;
  const size_t NQ     = (size_t)B_ * H_ * T_ * D_;

  __bf16* Wqkv_bf = (__bf16*)d_ws;
  __bf16* Wout_bf = Wqkv_bf + NW_QKV;
  __bf16* qT      = Wout_bf + NW_OUT;
  __bf16* kT      = qT + NQ;
  __bf16* vB      = kT + NQ;
  __bf16* xT      = vB + NQ;      // also reused as obuf
  __bf16* obuf    = xT;

  const int n8_qkv = (int)(NW_QKV / 8), n8_out = (int)(NW_OUT / 8);
  cvt_f32_bf16_kernel<<<(n8_qkv + 255) / 256, 256, 0, stream>>>(Wqkv, Wqkv_bf, n8_qkv);
  cvt_f32_bf16_kernel<<<(n8_out + 255) / 256, 256, 0, stream>>>(Wout, Wout_bf, n8_out);
  xpose_kernel<<<dim3(T_ / 64, C_ / 64, B_), 256, 0, stream>>>(x, xT);

  qkv_proj_kernel<<<dim3(T_ / 128, (3 * C_) / 128, B_), 256, 0, stream>>>(
      xT, Wqkv_bf, bqkv, qT, kT, vB);
  attn_kernel<<<dim3(T_ / 64, H_, B_), 128, 0, stream>>>(
      qT, kT, vB, mask, obuf);
  out_proj_kernel<<<dim3(T_ / 128, C_ / 128, B_), 256, 0, stream>>>(
      obuf, Wout_bf, bout, out);
}
